// DepthCueRectification_Sp_10909216932415
// MI455X (gfx1250) — compile-verified, matrix operand order checked by reading, not run on hardware
//
#include <hip/hip_runtime.h>

// ---------------------------------------------------------------------------
// DepthCueRectification for MI455X (gfx1250): all GEMMs via
// v_wmma_f32_16x16x32_bf16 (wave32). 8-wave blocks process a 32-row M-slab
// (2 M-tiles x 4 N-quadrants); wave pairs share B addresses through the WGP
// L0, halving L2 B-operand traffic (the true ceiling: all buffers are L2-
// resident in 192 MB). Each wave keeps 4 accumulators so the A operand and
// its f32->bf16 conversion amortize over 4 WMMAs. Attention (265 KB) and the
// MLP hidden tile (197 KB) stay fused in the 320 KB WGP LDS.
// ---------------------------------------------------------------------------

#define B_ 8
#define N_ 1024
#define DIM_ 768
#define DFF_ 3072
#define ROWS_ 1025   // N + CLS

typedef __attribute__((ext_vector_type(16))) __bf16 v16bf;
typedef __attribute__((ext_vector_type(8)))  float  v8f;

union BF16x16 {
  v16bf v;
  __bf16 h[16];
  unsigned short s[16];
  unsigned int   u[8];
};

__device__ __forceinline__ unsigned short f2bf(float f) {
  union { __bf16 b; unsigned short s; } u;
  u.b = (__bf16)f;                       // native v_cvt_pk_bf16_f32 when paired
  return u.s;
}

// A/B 16-bit operand striping (ISA 7.12.2): lane L (0..15) row M=L holds
// K = kb..kb+7 in halves 0..7 and K = kb+16..kb+23 in halves 8..15, where
// kb = k0 + (L>=16 ? 8 : 0).  Caller passes kb already including the hi-half.
__device__ __forceinline__ void load_bf16_op(BF16x16& a, const unsigned short* p, int kb) {
  const uint4 q0 = *(const uint4*)(p + kb);
  const uint4 q1 = *(const uint4*)(p + kb + 16);
  a.u[0] = q0.x; a.u[1] = q0.y; a.u[2] = q0.z; a.u[3] = q0.w;
  a.u[4] = q1.x; a.u[5] = q1.y; a.u[6] = q1.z; a.u[7] = q1.w;
}

__device__ __forceinline__ void load_f32_op(BF16x16& a, const float* p, int kb) {
  float4 f0 = *(const float4*)(p + kb);
  float4 f1 = *(const float4*)(p + kb + 4);
  float4 f2 = *(const float4*)(p + kb + 16);
  float4 f3 = *(const float4*)(p + kb + 20);
  a.h[0] = (__bf16)f0.x; a.h[1] = (__bf16)f0.y; a.h[2]  = (__bf16)f0.z; a.h[3]  = (__bf16)f0.w;
  a.h[4] = (__bf16)f1.x; a.h[5] = (__bf16)f1.y; a.h[6]  = (__bf16)f1.z; a.h[7]  = (__bf16)f1.w;
  a.h[8] = (__bf16)f2.x; a.h[9] = (__bf16)f2.y; a.h[10] = (__bf16)f2.z; a.h[11] = (__bf16)f2.w;
  a.h[12]= (__bf16)f3.x; a.h[13]= (__bf16)f3.y; a.h[14] = (__bf16)f3.z; a.h[15] = (__bf16)f3.w;
}

__device__ __forceinline__ v8f wmma_bf16(const BF16x16& a, const BF16x16& b, v8f c) {
  return __builtin_amdgcn_wmma_f32_16x16x32_bf16(false, a.v, false, b.v,
                                                 (short)0, c, false, false);
}

// ---------------------------------------------------------------------------
// Packing kernels: build bf16 B-operands (row = output column, K contiguous)
// ---------------------------------------------------------------------------
__global__ void k_pack_U(const float* __restrict__ U,
                         unsigned short* __restrict__ Ub,
                         unsigned short* __restrict__ UTb) {
  int i = blockIdx.x * blockDim.x + threadIdx.x;
  if (i >= DIM_ * DIM_) return;
  int r = i / DIM_, c = i % DIM_;
  unsigned short v = f2bf(U[i]);
  Ub[i] = v;                 // for t = xb @ U^T : B[d][n] = U[n][d]
  UTb[c * DIM_ + r] = v;     // for xs = (t*S) @ U : B[d][n] = U[d][n]
}

__global__ void k_pack_W1(const float* __restrict__ W1, unsigned short* __restrict__ W1T) {
  int i = blockIdx.x * blockDim.x + threadIdx.x;
  if (i >= 2 * DIM_ * DFF_) return;
  int r = i / DFF_, c = i % DFF_;
  W1T[(size_t)c * (2 * DIM_) + r] = f2bf(W1[i]);
}

__global__ void k_pack_W2(const float* __restrict__ W2, unsigned short* __restrict__ W2T) {
  int i = blockIdx.x * blockDim.x + threadIdx.x;
  if (i >= DFF_ * DIM_) return;
  int r = i / DIM_, c = i % DIM_;
  W2T[(size_t)c * DFF_ + r] = f2bf(W2[i]);
}

__global__ void k_pack_y(const float* __restrict__ y,
                         unsigned short* __restrict__ ybh,    // [B][N][D]
                         unsigned short* __restrict__ ybT,    // [B][D][N]
                         unsigned short* __restrict__ yfull)  // [B][ROWS][D]
{
  size_t i = (size_t)blockIdx.x * blockDim.x + threadIdx.x;
  if (i >= (size_t)B_ * ROWS_ * DIM_) return;
  int d = (int)(i % DIM_);
  int n = (int)((i / DIM_) % ROWS_);
  int b = (int)(i / ((size_t)DIM_ * ROWS_));
  unsigned short v = f2bf(y[i]);
  if (n == 0) {
    yfull[((size_t)b * ROWS_) * DIM_ + d] = v;               // CLS row passthrough
  } else {
    int m = n - 1;
    ybh[((size_t)b * N_ + m) * DIM_ + d] = v;
    ybT[((size_t)b * DIM_ + d) * N_ + m] = v;
  }
}

// ---------------------------------------------------------------------------
// pos = softmax(-|p_temp| * coords . pos_emb, axis=-1)   (batch independent)
// ---------------------------------------------------------------------------
__global__ __launch_bounds__(256) void k_pos(const float* __restrict__ coords,
                                             const float* __restrict__ pos_emb,
                                             const float* __restrict__ p_temp,
                                             float* __restrict__ pos) {
  __shared__ float buf[N_];
  __shared__ float red[256];
  int i = blockIdx.x, t = threadIdx.x;
  float pe[6];
#pragma unroll
  for (int c = 0; c < 6; ++c) pe[c] = pos_emb[i * 6 + c];
  float pt = fabsf(p_temp[0]);
  float lmax = -INFINITY;
  for (int j = t; j < N_; j += 256) {
    const float* cr = coords + ((size_t)i * N_ + j) * 6;
    float d = cr[0]*pe[0] + cr[1]*pe[1] + cr[2]*pe[2] +
              cr[3]*pe[3] + cr[4]*pe[4] + cr[5]*pe[5];
    float l = -pt * d;
    buf[j] = l;
    lmax = fmaxf(lmax, l);
  }
  red[t] = lmax; __syncthreads();
  for (int s = 128; s > 0; s >>= 1) { if (t < s) red[t] = fmaxf(red[t], red[t + s]); __syncthreads(); }
  float mx = red[0]; __syncthreads();
  float ls = 0.f;
  for (int j = t; j < N_; j += 256) { float e = __expf(buf[j] - mx); buf[j] = e; ls += e; }
  red[t] = ls; __syncthreads();
  for (int s = 128; s > 0; s >>= 1) { if (t < s) red[t] += red[t + s]; __syncthreads(); }
  float inv = 1.f / red[0];
  for (int j = t; j < N_; j += 256) pos[(size_t)i * N_ + j] = buf[j] * inv;
}

// ---------------------------------------------------------------------------
// t = xb @ U^T    (M = B*N, N = DIM, K = DIM)
// 8 waves / block: wave = (mt, quadrant); block = 32 rows x 256 cols
// ---------------------------------------------------------------------------
__global__ __launch_bounds__(256) void k_gemm_t(const float* __restrict__ x,
                                                const unsigned short* __restrict__ Ub,
                                                float* __restrict__ t) {
  int wave = threadIdx.x >> 5, lane = threadIdx.x & 31;
  int l15 = lane & 15, hi = lane >> 4;
  int mt = wave >> 2, wq = wave & 3;
  int rbase = blockIdx.x * 32 + mt * 16;
  int n0 = blockIdx.y * 256 + wq * 64;
  int r = rbase + l15;
  int b = r >> 10, m = r & (N_ - 1);
  const float* arow = x + ((size_t)b * ROWS_ + 1 + m) * DIM_;       // skip CLS
  const unsigned short* brow[4];
#pragma unroll
  for (int tt = 0; tt < 4; ++tt) brow[tt] = Ub + (size_t)(n0 + tt * 16 + l15) * DIM_;
  v8f acc[4] = {v8f{}, v8f{}, v8f{}, v8f{}};
  for (int k0 = 0; k0 < DIM_; k0 += 32) {
    int kb = k0 + hi * 8;
    BF16x16 A;
    load_f32_op(A, arow, kb);
    BF16x16 Bm[4];
#pragma unroll
    for (int tt = 0; tt < 4; ++tt) load_bf16_op(Bm[tt], brow[tt], kb);
#pragma unroll
    for (int tt = 0; tt < 4; ++tt) acc[tt] = wmma_bf16(A, Bm[tt], acc[tt]);
  }
#pragma unroll
  for (int tt = 0; tt < 4; ++tt) {
    int col = n0 + tt * 16 + l15;
#pragma unroll
    for (int v = 0; v < 8; ++v)
      t[(size_t)(rbase + v + hi * 8) * DIM_ + col] = acc[tt][v];
  }
}

// ---------------------------------------------------------------------------
// xs1 = (t*|S1|)@U ; xs2 = (t*|S2|)@U  — shared B operands, 8 accumulators
// ---------------------------------------------------------------------------
__global__ __launch_bounds__(256) void k_gemm_xs(const float* __restrict__ t,
                                                 const float* __restrict__ S1,
                                                 const float* __restrict__ S2,
                                                 const unsigned short* __restrict__ UTb,
                                                 unsigned short* __restrict__ xs1,
                                                 unsigned short* __restrict__ xs2) {
  int wave = threadIdx.x >> 5, lane = threadIdx.x & 31;
  int l15 = lane & 15, hi = lane >> 4;
  int mt = wave >> 2, wq = wave & 3;
  int rbase = blockIdx.x * 32 + mt * 16;
  int n0 = blockIdx.y * 256 + wq * 64;
  const float* arow = t + (size_t)(rbase + l15) * DIM_;
  const unsigned short* brow[4];
#pragma unroll
  for (int tt = 0; tt < 4; ++tt) brow[tt] = UTb + (size_t)(n0 + tt * 16 + l15) * DIM_;
  v8f a1[4] = {v8f{}, v8f{}, v8f{}, v8f{}};
  v8f a2[4] = {v8f{}, v8f{}, v8f{}, v8f{}};
  for (int k0 = 0; k0 < DIM_; k0 += 32) {
    int kb = k0 + hi * 8;
    BF16x16 A1, A2;
#pragma unroll
    for (int i = 0; i < 8; ++i) {
      float f0 = arow[kb + i], f1 = arow[kb + 16 + i];
      A1.h[i]     = (__bf16)(f0 * fabsf(S1[kb + i]));
      A1.h[8 + i] = (__bf16)(f1 * fabsf(S1[kb + 16 + i]));
      A2.h[i]     = (__bf16)(f0 * fabsf(S2[kb + i]));
      A2.h[8 + i] = (__bf16)(f1 * fabsf(S2[kb + 16 + i]));
    }
    BF16x16 Bm[4];
#pragma unroll
    for (int tt = 0; tt < 4; ++tt) load_bf16_op(Bm[tt], brow[tt], kb);
#pragma unroll
    for (int tt = 0; tt < 4; ++tt) {
      a1[tt] = wmma_bf16(A1, Bm[tt], a1[tt]);
      a2[tt] = wmma_bf16(A2, Bm[tt], a2[tt]);
    }
  }
#pragma unroll
  for (int tt = 0; tt < 4; ++tt) {
    int col = n0 + tt * 16 + l15;
#pragma unroll
    for (int v = 0; v < 8; ++v) {
      size_t o = (size_t)(rbase + v + hi * 8) * DIM_ + col;
      xs1[o] = f2bf(a1[tt][v]);
      xs2[o] = f2bf(a2[tt][v]);
    }
  }
}

// ---------------------------------------------------------------------------
// Fused attention per (batch, 32-row slab), 8 waves:
//   logits_k (WMMA) -> LDS -> softmax -> mix with pos -> entropy -> route
//   -> selected rows @ yb (WMMA) -> yfull (bf16) ; heat map to ws.
// Dynamic LDS: 2*32*1032 f32 attn + reductions ~ 265 KB (fits 320 KB WGP LDS).
// ---------------------------------------------------------------------------
#define ATT_LDR 1032

__global__ __launch_bounds__(256) void k_attn(const unsigned short* __restrict__ xs1,
                                              const unsigned short* __restrict__ xs2,
                                              const unsigned short* __restrict__ ybh,
                                              const unsigned short* __restrict__ ybT,
                                              const float* __restrict__ pos,
                                              const float* __restrict__ gating,
                                              const float* __restrict__ h_temp,
                                              unsigned short* __restrict__ yfull,
                                              float* __restrict__ heat) {
  extern __shared__ float smem[];
  float* attn = smem;                         // [2][32][ATT_LDR]
  float* red  = smem + 2 * 32 * ATT_LDR;      // [64][4]
  float* ent  = red + 256;                    // [64]
  int*   sel  = (int*)(ent + 64);             // [32]

  int tid = threadIdx.x, wave = tid >> 5, lane = tid & 31;
  int l15 = lane & 15, hi = lane >> 4;
  int mt = wave >> 2, wq = wave & 3;
  int it = blockIdx.x, b = blockIdx.y;
  int ibase = it * 32;
  const float scale = 0.03608439182435161f;   // DIM^-0.5

  // ---- Phase A: logits for both heads -> LDS (wave strip = 16x64 per pass) ----
  for (int kk = 0; kk < 2; ++kk) {
    const unsigned short* xs = kk ? xs2 : xs1;
    const unsigned short* arow = xs + ((size_t)b * N_ + ibase + mt * 16 + l15) * DIM_;
    for (int n0 = wq * 64; n0 < N_; n0 += 256) {
      const unsigned short* brow[4];
#pragma unroll
      for (int tt = 0; tt < 4; ++tt)
        brow[tt] = ybh + ((size_t)b * N_ + n0 + tt * 16 + l15) * DIM_;
      v8f acc[4] = {v8f{}, v8f{}, v8f{}, v8f{}};
      for (int k0 = 0; k0 < DIM_; k0 += 32) {
        int kb = k0 + hi * 8;
        BF16x16 A;
        load_bf16_op(A, arow, kb);
        BF16x16 Bm[4];
#pragma unroll
        for (int tt = 0; tt < 4; ++tt) load_bf16_op(Bm[tt], brow[tt], kb);
#pragma unroll
        for (int tt = 0; tt < 4; ++tt) acc[tt] = wmma_bf16(A, Bm[tt], acc[tt]);
      }
#pragma unroll
      for (int tt = 0; tt < 4; ++tt) {
#pragma unroll
        for (int v = 0; v < 8; ++v) {
          int row = mt * 16 + v + hi * 8;
          attn[(kk * 32 + row) * ATT_LDR + n0 + tt * 16 + l15] = acc[tt][v] * scale;
        }
      }
    }
  }
  __syncthreads();

  // ---- Phase B: softmax, mix with pos, entropy (4 threads per row, 64 rows) ----
  float g  = 1.f / (1.f + __expf(-gating[0]));
  float ht = h_temp[0];
  {
    int rr = tid >> 2;          // 0..63 : (head, row)
    int q  = tid & 3;
    int m  = rr & 31;
    float* row = attn + rr * ATT_LDR;
    int j0 = q * 256, j1 = j0 + 256;

    float mx = -INFINITY;
    for (int j = j0; j < j1; ++j) mx = fmaxf(mx, row[j]);
    red[rr * 4 + q] = mx; __syncthreads();
    mx = fmaxf(fmaxf(red[rr*4], red[rr*4+1]), fmaxf(red[rr*4+2], red[rr*4+3]));
    __syncthreads();

    float s = 0.f;
    for (int j = j0; j < j1; ++j) { float e = __expf(row[j] - mx); row[j] = e; s += e; }
    red[rr * 4 + q] = s; __syncthreads();
    s = red[rr*4] + red[rr*4+1] + red[rr*4+2] + red[rr*4+3];
    __syncthreads();

    float c1 = (1.f - g) / s;
    const float* prow = pos + (size_t)(ibase + m) * N_;
    float ep = 0.f;
    for (int j = j0; j < j1; ++j) {
      float a = c1 * row[j] + g * prow[j];
      row[j] = a;
      ep += -a * __logf(a + 1e-8f);
    }
    red[rr * 4 + q] = ep; __syncthreads();
    if (q == 0) ent[rr] = red[rr*4] + red[rr*4+1] + red[rr*4+2] + red[rr*4+3];
  }
  __syncthreads();

  if (tid < 32) {
    float e0 = ent[tid], e1 = ent[32 + tid];
    float h0 = 2.f - 2.f / (1.f + __expf(-ht * e0));
    float h1 = 2.f - 2.f / (1.f + __expf(-ht * e1));
    int s = (h0 >= h1) ? 0 : 1;                  // fgbg route
    sel[tid] = s;
    heat[(size_t)b * N_ + ibase + tid] = s ? h1 : h0;
  }
  __syncthreads();

  // ---- Phase C: y_out = attn_sel @ yb  (32 x 768, K = 1024) ----
  int am = mt * 16 + l15;
  const float* arow = attn + (sel[am] * 32 + am) * ATT_LDR;
  for (int n0 = wq * 64; n0 < DIM_; n0 += 256) {
    const unsigned short* brow[4];
#pragma unroll
    for (int tt = 0; tt < 4; ++tt)
      brow[tt] = ybT + ((size_t)b * DIM_ + n0 + tt * 16 + l15) * N_;
    v8f acc[4] = {v8f{}, v8f{}, v8f{}, v8f{}};
    for (int k0 = 0; k0 < N_; k0 += 32) {
      int kb = k0 + hi * 8;
      BF16x16 A;
      load_f32_op(A, arow, kb);     // LDS f32 -> bf16 in-register (amortized 4x)
      BF16x16 Bm[4];
#pragma unroll
      for (int tt = 0; tt < 4; ++tt) load_bf16_op(Bm[tt], brow[tt], kb);
#pragma unroll
      for (int tt = 0; tt < 4; ++tt) acc[tt] = wmma_bf16(A, Bm[tt], acc[tt]);
    }
#pragma unroll
    for (int tt = 0; tt < 4; ++tt) {
      int col = n0 + tt * 16 + l15;
#pragma unroll
      for (int v = 0; v < 8; ++v) {
        int row = mt * 16 + v + hi * 8;
        yfull[((size_t)b * ROWS_ + 1 + ibase + row) * DIM_ + col] = f2bf(acc[tt][v]);
      }
    }
  }
}

// ---------------------------------------------------------------------------
// Fused MLP per (batch, 32-row slab), 8 waves:
//   h = [x, yfull] @ W1 + b1 ; GELU -> LDS bf16 (32x3072) ; @ W2 + b2
//   out = x + heat * xp   (CLS row unscaled)
// ---------------------------------------------------------------------------
#define H_LDR 3080

__global__ __launch_bounds__(256) void k_mlp(const float* __restrict__ x,
                                             const unsigned short* __restrict__ yfull,
                                             const unsigned short* __restrict__ W1T,
                                             const unsigned short* __restrict__ W2T,
                                             const float* __restrict__ b1,
                                             const float* __restrict__ b2,
                                             const float* __restrict__ heat,
                                             float* __restrict__ out) {
  extern __shared__ unsigned short hs[];      // [32][H_LDR] bf16
  int tid = threadIdx.x, wave = tid >> 5, lane = tid & 31;
  int l15 = lane & 15, hi = lane >> 4;
  int mt = wave >> 2, wq = wave & 3;
  int it = blockIdx.x, b = blockIdx.y;
  int nbase = it * 32;
  int na = nbase + mt * 16 + l15;
  if (na > ROWS_ - 1) na = ROWS_ - 1;          // clamp partial slab
  const float* xrow = x + ((size_t)b * ROWS_ + na) * DIM_;
  const unsigned short* yrow = yfull + ((size_t)b * ROWS_ + na) * DIM_;

  // GEMM1 + bias + GELU -> LDS
  for (int n0 = wq * 64; n0 < DFF_; n0 += 256) {
    const unsigned short* brow[4];
#pragma unroll
    for (int tt = 0; tt < 4; ++tt)
      brow[tt] = W1T + (size_t)(n0 + tt * 16 + l15) * (2 * DIM_);
    v8f acc[4] = {v8f{}, v8f{}, v8f{}, v8f{}};
    for (int k0 = 0; k0 < 2 * DIM_; k0 += 32) {
      int kb = k0 + hi * 8;
      BF16x16 A;
      if (kb < DIM_) load_f32_op(A, xrow, kb);          // x half of concat
      else           load_bf16_op(A, yrow, kb - DIM_);  // y_full half
      BF16x16 Bm[4];
#pragma unroll
      for (int tt = 0; tt < 4; ++tt) load_bf16_op(Bm[tt], brow[tt], kb);
#pragma unroll
      for (int tt = 0; tt < 4; ++tt) acc[tt] = wmma_bf16(A, Bm[tt], acc[tt]);
    }
#pragma unroll
    for (int tt = 0; tt < 4; ++tt) {
      int col = n0 + tt * 16 + l15;
      float bb = b1[col];
#pragma unroll
      for (int v = 0; v < 8; ++v) {
        int row = mt * 16 + v + hi * 8;
        float hv = acc[tt][v] + bb;
        float gl = 0.5f * hv * (1.f + erff(hv * 0.70710678118f));   // exact GELU
        hs[row * H_LDR + col] = f2bf(gl);
      }
    }
  }
  __syncthreads();

  // GEMM2 + bias + heat/residual epilogue
  const unsigned short* arow = hs + (mt * 16 + l15) * H_LDR;
  for (int n0 = wq * 64; n0 < DIM_; n0 += 256) {
    const unsigned short* brow[4];
#pragma unroll
    for (int tt = 0; tt < 4; ++tt)
      brow[tt] = W2T + (size_t)(n0 + tt * 16 + l15) * DFF_;
    v8f acc[4] = {v8f{}, v8f{}, v8f{}, v8f{}};
    for (int k0 = 0; k0 < DFF_; k0 += 32) {
      int kb = k0 + hi * 8;
      BF16x16 A;
      load_bf16_op(A, arow, kb);     // ds_load_b128 from LDS
      BF16x16 Bm[4];
#pragma unroll
      for (int tt = 0; tt < 4; ++tt) load_bf16_op(Bm[tt], brow[tt], kb);
#pragma unroll
      for (int tt = 0; tt < 4; ++tt) acc[tt] = wmma_bf16(A, Bm[tt], acc[tt]);
    }
#pragma unroll
    for (int tt = 0; tt < 4; ++tt) {
      int col = n0 + tt * 16 + l15;
      float bb = b2[col];
#pragma unroll
      for (int v = 0; v < 8; ++v) {
        int row = mt * 16 + v + hi * 8;
        int n = nbase + row;
        if (n < ROWS_) {
          float xp = acc[tt][v] + bb;
          float hm = (n == 0) ? 1.f : heat[(size_t)b * N_ + (n - 1)];
          size_t o = ((size_t)b * ROWS_ + n) * DIM_ + col;
          out[o] = x[o] + hm * xp;
        }
      }
    }
  }
}

// ---------------------------------------------------------------------------
extern "C" void kernel_launch(void* const* d_in, const int* in_sizes, int n_in,
                              void* d_out, int out_size, void* d_ws, size_t ws_size,
                              hipStream_t stream) {
  const float* x       = (const float*)d_in[0];
  const float* y       = (const float*)d_in[1];
  const float* coords  = (const float*)d_in[2];
  const float* U       = (const float*)d_in[3];
  const float* S1      = (const float*)d_in[4];
  const float* S2      = (const float*)d_in[5];
  const float* gating  = (const float*)d_in[6];
  const float* h_temp  = (const float*)d_in[7];
  const float* p_temp  = (const float*)d_in[8];
  const float* pos_emb = (const float*)d_in[9];
  const float* W1      = (const float*)d_in[10];
  const float* b1      = (const float*)d_in[11];
  const float* W2      = (const float*)d_in[12];
  const float* b2      = (const float*)d_in[13];
  float* out = (float*)d_out;

  // ---- workspace layout (all sizes are multiples of 256 bytes; ~104 MB) ----
  char* base = (char*)d_ws;
  size_t off = 0;
  auto take = [&](size_t bytes) { char* p = base + off; off += (bytes + 255) & ~(size_t)255; return p; };
  float*          pos    = (float*)         take((size_t)N_ * N_ * 4);
  unsigned short* Ub     = (unsigned short*)take((size_t)DIM_ * DIM_ * 2);
  unsigned short* UTb    = (unsigned short*)take((size_t)DIM_ * DIM_ * 2);
  float*          tbuf   = (float*)         take((size_t)B_ * N_ * DIM_ * 4);
  unsigned short* xs1h   = (unsigned short*)take((size_t)B_ * N_ * DIM_ * 2);
  unsigned short* xs2h   = (unsigned short*)take((size_t)B_ * N_ * DIM_ * 2);
  unsigned short* ybh    = (unsigned short*)take((size_t)B_ * N_ * DIM_ * 2);
  unsigned short* ybT    = (unsigned short*)take((size_t)B_ * DIM_ * N_ * 2);
  unsigned short* yfull  = (unsigned short*)take((size_t)B_ * ROWS_ * DIM_ * 2);
  unsigned short* W1T    = (unsigned short*)take((size_t)DFF_ * 2 * DIM_ * 2);
  unsigned short* W2T    = (unsigned short*)take((size_t)DIM_ * DFF_ * 2);
  float*          heat   = (float*)         take((size_t)B_ * N_ * 4);
  (void)ws_size; (void)in_sizes; (void)n_in; (void)out_size;

  // large dynamic LDS (attention ~265 KB, MLP ~197 KB; WGP has 320 KB)
  const size_t attn_smem = (size_t)(2 * 32 * ATT_LDR + 256 + 64) * 4 + 32 * 4;
  const size_t mlp_smem  = (size_t)32 * H_LDR * 2;
  (void)hipFuncSetAttribute((const void*)k_attn, hipFuncAttributeMaxDynamicSharedMemorySize, (int)attn_smem);
  (void)hipFuncSetAttribute((const void*)k_mlp,  hipFuncAttributeMaxDynamicSharedMemorySize, (int)mlp_smem);

  // ---- packing + pos (independent) ----
  k_pack_U <<<dim3((DIM_*DIM_ + 255)/256),        256, 0, stream>>>(U, Ub, UTb);
  k_pack_W1<<<dim3((2*DIM_*DFF_ + 255)/256),      256, 0, stream>>>(W1, W1T);
  k_pack_W2<<<dim3((DFF_*DIM_ + 255)/256),        256, 0, stream>>>(W2, W2T);
  k_pack_y <<<dim3((B_*ROWS_*DIM_ + 255)/256),    256, 0, stream>>>(y, ybh, ybT, yfull);
  k_pos    <<<dim3(N_),                           256, 0, stream>>>(coords, pos_emb, p_temp, pos);

  // ---- t = xb @ U^T ----
  k_gemm_t <<<dim3((B_*N_)/32, DIM_/256), 256, 0, stream>>>(x, Ub, tbuf);

  // ---- xs1/xs2 = (t*|S|) @ U ----
  k_gemm_xs<<<dim3((B_*N_)/32, DIM_/256), 256, 0, stream>>>(tbuf, S1, S2, UTb, xs1h, xs2h);

  // ---- fused attention (32-row slabs) ----
  k_attn<<<dim3(N_/32, B_), 256, attn_smem, stream>>>(xs1h, xs2h, ybh, ybT, pos,
                                                      gating, h_temp, yfull, heat);

  // ---- fused MLP + epilogue (32-row slabs) ----
  k_mlp<<<dim3((ROWS_ + 31)/32, B_), 256, mlp_smem, stream>>>(x, yfull, W1T, W2T,
                                                              b1, b2, heat, out);
}
